// CausalAttention_12644383719757
// MI455X (gfx1250) — matrix operand order repaired and checked by reading, not compiled, not run on hardware
//
#include <hip/hip_runtime.h>
#include <hip/hip_bf16.h>

typedef __attribute__((ext_vector_type(16))) _Float16 v16h;
typedef __attribute__((ext_vector_type(8)))  _Float16 v8h;
typedef __attribute__((ext_vector_type(8)))  float    v8f;

// ---------------------------------------------------------------------------
// WMMA fragment loaders (CDNA5 v_wmma_f32_16x16x32_f16 layouts, wave32)
// A (16x32, MxK): lane L<16 -> M=L, halves 0..7 = K 0..7, halves 8..15 = K 16..23
//                 lane L>=16 -> M=L-16, K offset +8 within each 16-chunk.
// B (32x16, KxN): lane n -> N=n%16, same K layout as A (halves 0..7 = K 0..7 / 8..15, halves 8..15 = K 16..23 / 24..31)
// => with A row-major [M][K] and B stored transposed Bt[N][K], all loads are
//    contiguous 16B / 32B per lane.
// ---------------------------------------------------------------------------
__device__ __forceinline__ v16h frag_a(const _Float16* A, int lda, int m0, int k0, int lane) {
  const _Float16* p = A + (size_t)(m0 + (lane & 15)) * lda + k0 + ((lane >> 4) << 3);
  v8h lo = *(const v8h*)p;
  v8h hi = *(const v8h*)(p + 16);
  v16h r;
#pragma unroll
  for (int i = 0; i < 8; ++i) { r[i] = lo[i]; r[i + 8] = hi[i]; }
  return r;
}

__device__ __forceinline__ v16h frag_b(const _Float16* Bt, int ldb, int n0, int k0, int lane) {
  const _Float16* p = Bt + (size_t)(n0 + (lane & 15)) * ldb + k0 + ((lane >> 4) << 3);
  v8h lo = *(const v8h*)p;
  v8h hi = *(const v8h*)(p + 16);
  v16h r;
#pragma unroll
  for (int i = 0; i < 8; ++i) { r[i] = lo[i]; r[i + 8] = hi[i]; }
  return r;
}

__device__ __forceinline__ v8f wmma32(v16h a, v16h b, v8f c) {
  return __builtin_amdgcn_wmma_f32_16x16x32_f16(false, a, false, b, (short)0, c,
                                                false, false);
}

// wave-local LDS fence (waves in a block run different trip counts -> no s_barrier)
__device__ __forceinline__ void lds_sync() {
  asm volatile("s_wait_dscnt 0" ::: "memory");
}

// ---------------------------------------------------------------------------
// f32 -> f16 conversion / transposed conversion
// ---------------------------------------------------------------------------
__global__ void cvt_f16_kernel(const float* __restrict__ src, _Float16* __restrict__ dst, int n) {
  int i = blockIdx.x * blockDim.x + threadIdx.x;
  if (i < n) dst[i] = (_Float16)src[i];
}

// w [K][N] f32 -> wT [N][K] f16
__global__ void cvt_transpose_kernel(const float* __restrict__ w, _Float16* __restrict__ wT,
                                     int K, int N) {
  int i = blockIdx.x * blockDim.x + threadIdx.x;
  if (i < K * N) {
    int k = i / N, n = i % N;
    wT[(size_t)n * K + k] = (_Float16)w[i];
  }
}

// ---------------------------------------------------------------------------
// GEMM: C[M][N](f32) = A[M][K](f16, row-major) * Bt[N][K](f16)^T + bias[N]
// 64x64 per wave, 8 waves/block (2x4) -> 128x256 per block. M%128==0, N%256==0, K%32==0.
// ---------------------------------------------------------------------------
__global__ __launch_bounds__(256) void gemm_wmma_kernel(const _Float16* __restrict__ A,
                                                        const _Float16* __restrict__ Bt,
                                                        const float* __restrict__ bias,
                                                        float* __restrict__ C,
                                                        int M, int N, int K) {
  int lane = threadIdx.x & 31;
  int wid  = threadIdx.x >> 5;
  int m0 = blockIdx.x * 128 + (wid & 1) * 64;
  int n0 = blockIdx.y * 256 + (wid >> 1) * 64;

  v8f acc[4][4] = {};
  for (int k0 = 0; k0 < K; k0 += 32) {
    v16h a[4], b[4];
#pragma unroll
    for (int i = 0; i < 4; ++i) a[i] = frag_a(A, K, m0 + 16 * i, k0, lane);
#pragma unroll
    for (int j = 0; j < 4; ++j) b[j] = frag_b(Bt, K, n0 + 16 * j, k0, lane);
#pragma unroll
    for (int i = 0; i < 4; ++i)
#pragma unroll
      for (int j = 0; j < 4; ++j)
        acc[i][j] = wmma32(a[i], b[j], acc[i][j]);
  }

  int hi = lane >> 4, nl = lane & 15;
#pragma unroll
  for (int j = 0; j < 4; ++j) {
    int col = n0 + 16 * j + nl;
    float bj = bias ? bias[col] : 0.0f;
#pragma unroll
    for (int i = 0; i < 4; ++i) {
#pragma unroll
      for (int r = 0; r < 8; ++r) {
        int row = m0 + 16 * i + r + 8 * hi;
        C[(size_t)row * N + col] = acc[i][j][r] + bj;
      }
    }
  }
}

// ---------------------------------------------------------------------------
// RoPE + repack: qkv f32 [B,T,3C] -> q16,k16 [B,H,T,D] (rotated), vt16 [B,H,D,T]
// One thread per (b,h,t, dim-pair).
// ---------------------------------------------------------------------------
__global__ void rope_pack_kernel(const float* __restrict__ qkv,
                                 _Float16* __restrict__ q16,
                                 _Float16* __restrict__ k16,
                                 _Float16* __restrict__ vt16,
                                 int B, int T, int H, int D) {
  int idx = blockIdx.x * blockDim.x + threadIdx.x;
  int Dh = D / 2;
  int total = B * H * T * Dh;
  if (idx >= total) return;
  int j = idx % Dh;
  int t = (idx / Dh) % T;
  int h = (idx / (Dh * T)) % H;
  int b = idx / (Dh * T * H);
  int C = H * D;

  const float* p = qkv + (size_t)(b * T + t) * (3 * C) + h * D + 2 * j;
  float q0 = p[0], q1 = p[1];
  float k0 = p[C], k1 = p[C + 1];
  float v0 = p[2 * C], v1 = p[2 * C + 1];

  // inv_freq = theta^{-(2j)/D}; ln(10000) = 9.210340371976184
  float inv = __expf(-((float)(2 * j) / (float)D) * 9.210340371976184f);
  float ang = (float)t * inv;
  float c = __cosf(ang), s = __sinf(ang);

  size_t oq = ((size_t)(b * H + h) * T + t) * D + 2 * j;
  q16[oq]     = (_Float16)(q0 * c - q1 * s);
  q16[oq + 1] = (_Float16)(q1 * c + q0 * s);
  k16[oq]     = (_Float16)(k0 * c - k1 * s);
  k16[oq + 1] = (_Float16)(k1 * c + k0 * s);

  size_t ov = ((size_t)(b * H + h) * D + 2 * j) * T + t;  // [b][h][d][t]
  vt16[ov]     = (_Float16)v0;
  vt16[ov + T] = (_Float16)v1;
}

// ---------------------------------------------------------------------------
// Flash-style causal attention. One wave per 16-query tile of one (b,h).
// S = (Q Kt) * 1/8 via WMMA; online softmax by lanes 0..15 through per-wave LDS;
// P staged in LDS as f16, reloaded as A-fragments; Y += P V via WMMA on Vt[D][T].
// Output yatt f16 [B, T, H*D] (row-major A for the projection GEMM).
// ---------------------------------------------------------------------------
__global__ __launch_bounds__(256) void attn_wmma_kernel(const _Float16* __restrict__ q16,
                                                        const _Float16* __restrict__ k16,
                                                        const _Float16* __restrict__ vt16,
                                                        _Float16* __restrict__ yatt,
                                                        int B, int H, int T) {
  const int D = 64;
  __shared__ float    sS[8][16 * 32];
  __shared__ _Float16 sP[8][16 * 32];
  __shared__ float    sR[8][16];

  int lane = threadIdx.x & 31;
  int wid  = threadIdx.x >> 5;
  int QT = T >> 4;
  int gw = blockIdx.x * 8 + wid;
  int qt = gw % QT;
  int h  = (gw / QT) % H;
  int b  = gw / (QT * H);
  int m0 = qt * 16;

  const _Float16* Q  = q16  + (size_t)(b * H + h) * T * D;
  const _Float16* Kh = k16  + (size_t)(b * H + h) * T * D;
  const _Float16* Vt = vt16 + (size_t)(b * H + h) * D * T;

  float*    Ss = sS[wid];
  _Float16* Ps = sP[wid];
  float*    Rs = sR[wid];

  v16h qa0 = frag_a(Q, D, m0, 0, lane);
  v16h qa1 = frag_a(Q, D, m0, 32, lane);
  v8f yacc[4] = {};
  float m_i = -1e30f, l_i = 0.0f;  // live in lanes 0..15
  int hi = lane >> 4, nl = lane & 15;

  for (int kk = 0; kk < m0 + 16; kk += 32) {
    // --- S = Q Kt over a 16x32 key block (two 16x16 tiles) ---
    v8f s0 = {}, s1 = {};
    s0 = wmma32(qa0, frag_b(Kh, D, kk,      0,  lane), s0);
    s0 = wmma32(qa1, frag_b(Kh, D, kk,      32, lane), s0);
    s1 = wmma32(qa0, frag_b(Kh, D, kk + 16, 0,  lane), s1);
    s1 = wmma32(qa1, frag_b(Kh, D, kk + 16, 32, lane), s1);
#pragma unroll
    for (int r = 0; r < 8; ++r) {
      int row = r + 8 * hi;
      Ss[row * 32 + nl]      = s0[r] * 0.125f;
      Ss[row * 32 + 16 + nl] = s1[r] * 0.125f;
    }
    lds_sync();

    // --- online softmax: lane r owns query row m0+r ---
    if (lane < 16) {
      int q = m0 + lane;
      float sv[32];
      float rmax = -1e30f;
#pragma unroll
      for (int j = 0; j < 32; ++j) {
        float v = Ss[lane * 32 + j];
        if (kk + j > q) v = -1e30f;  // causal mask
        sv[j] = v;
        rmax = fmaxf(rmax, v);
      }
      float mnew  = fmaxf(m_i, rmax);
      float alpha = __expf(m_i - mnew);
      float psum  = 0.0f;
#pragma unroll
      for (int j = 0; j < 32; ++j) {
        float pv = __expf(sv[j] - mnew);
        psum += pv;
        Ps[lane * 32 + j] = (_Float16)pv;
      }
      l_i = l_i * alpha + psum;
      m_i = mnew;
      Rs[lane] = alpha;
    }
    lds_sync();

    // --- rescale running Y by alpha[row] ---
#pragma unroll
    for (int r = 0; r < 8; ++r) {
      float alpha = Rs[r + 8 * hi];
#pragma unroll
      for (int jt = 0; jt < 4; ++jt) yacc[jt][r] *= alpha;
    }

    // --- Y += P V : P from LDS as A-frag, Vt[D][T] gives contiguous B-frags ---
    v16h pa = frag_a(Ps, 32, 0, 0, lane);
#pragma unroll
    for (int jt = 0; jt < 4; ++jt) {
      v16h vb = frag_b(Vt, T, jt * 16, kk, lane);
      yacc[jt] = wmma32(pa, vb, yacc[jt]);
    }
  }

  if (lane < 16) Rs[lane] = 1.0f / l_i;
  lds_sync();

#pragma unroll
  for (int r = 0; r < 8; ++r) {
    int row = r + 8 * hi;
    float rinv = Rs[row];
    int t = m0 + row;
    _Float16* o = yatt + (size_t)(b * T + t) * (H * D) + h * D;
#pragma unroll
    for (int jt = 0; jt < 4; ++jt)
      o[jt * 16 + nl] = (_Float16)(yacc[jt][r] * rinv);
  }
}

// ---------------------------------------------------------------------------
extern "C" void kernel_launch(void* const* d_in, const int* in_sizes, int n_in,
                              void* d_out, int out_size, void* d_ws, size_t ws_size,
                              hipStream_t stream) {
  const float* x      = (const float*)d_in[0];
  const float* w_attn = (const float*)d_in[1];
  const float* b_attn = (const float*)d_in[2];
  const float* w_proj = (const float*)d_in[3];
  const float* b_proj = (const float*)d_in[4];
  float* out = (float*)d_out;

  const int B = 2, T = 2048, C = 1024, H = 16, D = 64;
  const int M = B * T;  // 4096

  char* ws = (char*)d_ws;
  // workspace layout (yatt reuses the x16 region once the QKV GEMM is done)
  _Float16* x16    = (_Float16*)(ws + 0);            //  8 MB
  _Float16* yatt   = (_Float16*)(ws + 0);            //  8 MB (after gemm1)
  _Float16* wattnT = (_Float16*)(ws + 8388608);      //  6 MB
  _Float16* wprojT = (_Float16*)(ws + 14680064);     //  2 MB
  float*    qkv32  = (float*)   (ws + 16777216);     // 48 MB
  _Float16* q16    = (_Float16*)(ws + 67108864);     //  8 MB
  _Float16* k16    = (_Float16*)(ws + 75497472);     //  8 MB
  _Float16* vt16   = (_Float16*)(ws + 83886080);     //  8 MB (ends 92274688)

  int nx = M * C;
  cvt_f16_kernel<<<(nx + 255) / 256, 256, 0, stream>>>(x, x16, nx);
  cvt_transpose_kernel<<<(C * 3 * C + 255) / 256, 256, 0, stream>>>(w_attn, wattnT, C, 3 * C);
  cvt_transpose_kernel<<<(C * C + 255) / 256, 256, 0, stream>>>(w_proj, wprojT, C, C);

  // qkv = x @ w_attn + b_attn
  dim3 g1(M / 128, (3 * C) / 256);
  gemm_wmma_kernel<<<g1, 256, 0, stream>>>(x16, wattnT, b_attn, qkv32, M, 3 * C, C);

  // RoPE + head repack (+ V transpose)
  int nrope = B * H * T * (D / 2);
  rope_pack_kernel<<<(nrope + 255) / 256, 256, 0, stream>>>(qkv32, q16, k16, vt16, B, T, H, D);

  // causal flash attention
  int waves = B * H * (T / 16);  // 4096 waves, 8 per block
  attn_wmma_kernel<<<waves / 8, 256, 0, stream>>>(q16, k16, vt16, yatt, B, H, T);

  // out = yatt @ w_proj + b_proj
  dim3 g2(M / 128, C / 256);
  gemm_wmma_kernel<<<g2, 256, 0, stream>>>(yatt, wprojT, b_proj, out, M, C, C);
}
